// LocalEncoder_4320737100314
// MI455X (gfx1250) — matrix operand order repaired and checked by reading, not compiled, not run on hardware
//
#include <hip/hip_runtime.h>

typedef __attribute__((ext_vector_type(16))) _Float16 v16h;
typedef __attribute__((ext_vector_type(8)))  _Float16 h8;
typedef __attribute__((ext_vector_type(8)))  float    v8f;
typedef __attribute__((ext_vector_type(4)))  int      i4;

#define B_    4
#define L_    2048
#define D_    768
#define H_    12
#define DH_   64
#define P_    512
#define BL_   (B_ * L_)
#define VHASH_ 50000
#define SCALE_ 0.125f  /* 1/sqrt(64) */

#define GAS __attribute__((address_space(1)))
#define LAS __attribute__((address_space(3)))

#if __has_builtin(__builtin_amdgcn_global_load_async_to_lds_b128)
#define HAVE_ASYNC 1
#else
#define HAVE_ASYNC 0
#endif

#if HAVE_ASYNC
#if __has_builtin(__builtin_amdgcn_s_wait_asynccnt)
#define ASYNC_WAIT(n) __builtin_amdgcn_s_wait_asynccnt(n)
#else
#define ASYNC_WAIT(n) asm volatile("s_wait_asynccnt %0" ::"i"(n) : "memory")
#endif
#else
#define ASYNC_WAIT(n) ((void)0)
#endif

static __device__ __forceinline__ v16h make_v16h(h8 lo, h8 hi) {
  union { v16h v; h8 h[2]; } u;
  u.h[0] = lo; u.h[1] = hi;
  return u.v;
}

// ---------------------------------------------------------------------------
// Weight convert + transpose: W f32 [N,K] -> Wt f16 [K,N]
// ---------------------------------------------------------------------------
__global__ void cvt_t_kernel(const float* __restrict__ W, _Float16* __restrict__ Wt,
                             int N, int K) {
  long i = (long)blockIdx.x * 256 + threadIdx.x;
  long tot = (long)N * K;
  if (i < tot) {
    int n = (int)(i / K);
    int k = (int)(i % K);
    Wt[(long)k * N + n] = (_Float16)W[i];
  }
}

// ---------------------------------------------------------------------------
// Byte-embedding gather + rolling-hash ngram gather into concat f16 buffer
// ---------------------------------------------------------------------------
__global__ void __launch_bounds__(256)
gather_kernel(const int* __restrict__ bytes, const float* __restrict__ byte_emb,
              const float* __restrict__ ngram_tab, float* __restrict__ x0,
              _Float16* __restrict__ cat) {
  int row = blockIdx.x;              // b*L + l
  int b = row / L_, l = row % L_;
  __shared__ int hsh[6];
  __shared__ int vld[6];
  if (threadIdx.x < 6) {
    int n = (int)threadIdx.x + 3;
    int c = n >> 1;
    int i = l - c;
    int ok = (i >= 0) && (i <= L_ - n);
    int h = 0;
    if (ok) {
      for (int j = 0; j < n; ++j)
        h = (h * 256 + bytes[b * L_ + i + j]) % VHASH_;
    }
    hsh[threadIdx.x] = h;
    vld[threadIdx.x] = ok;
  }
  __syncthreads();
  int bv = bytes[row];
  for (int d = threadIdx.x; d < D_; d += 256)
    x0[(long)row * D_ + d] = byte_emb[(long)bv * D_ + d];
  for (int c = threadIdx.x; c < 6 * D_; c += 256) {
    int idx = c / D_, d = c % D_;
    float v = vld[idx] ? ngram_tab[((long)idx * VHASH_ + hsh[idx]) * D_ + d] : 0.0f;
    cat[(long)row * (6 * D_) + c] = (_Float16)v;
  }
}

// ---------------------------------------------------------------------------
// WMMA GEMM:  out = act(A[M,K](f16) @ Bt[K,N](f16)  + bias + residual)
// Block tile 64x128, 8 waves, wave = 16x64 (4 accumulators), K step 32.
// Double-buffered LDS staging via async global->LDS copies (ASYNCcnt).
// ---------------------------------------------------------------------------
#define LDA_P 40    // halfs per A-tile row (32 + 8 pad)
#define LDB_P 136   // halfs per Bt-tile row (128 + 8 pad)

__global__ void __launch_bounds__(256)
gemm_wmma_kernel(const _Float16* __restrict__ A, const _Float16* __restrict__ Bt,
                 const float* __restrict__ bias, const float* __restrict__ resid,
                 float* __restrict__ Cf, _Float16* __restrict__ Ch,
                 int M, int N, int K, int do_relu) {
  __shared__ __align__(16) _Float16 sA[2][64 * LDA_P];
  __shared__ __align__(16) _Float16 sB[2][32 * LDB_P];

  int tid  = threadIdx.x;
  int wave = tid >> 5, lane = tid & 31;
  int m0 = blockIdx.y * 64;
  int n0 = blockIdx.x * 128;
  int wm = wave >> 1;          // 0..3 : row tile of 16
  int wn = wave & 1;           // 0..1 : 64-col half
  int half_sel = lane >> 4;
  int mrow = lane & 15;

  v8f acc[4] = {};

  int a_row = tid >> 2;            // 0..63
  int a_col = (tid & 3) * 8;       // 0..24
  int b_row = tid >> 3;            // 0..31
  int b_col = (tid & 7) * 16;      // 0..112

  // stage tile tt into LDS buffer bufi (3 x 128-bit per thread)
  auto stage = [&](int tt, int bufi) {
    int k0 = tt * 32;
    const _Float16* ga  = &A[(long)(m0 + a_row) * K + k0 + a_col];
    const _Float16* gb0 = &Bt[(long)(k0 + b_row) * N + n0 + b_col];
    _Float16* la  = &sA[bufi][a_row * LDA_P + a_col];
    _Float16* lb0 = &sB[bufi][b_row * LDB_P + b_col];
#if HAVE_ASYNC
    __builtin_amdgcn_global_load_async_to_lds_b128(
        (GAS i4*)(void*)ga, (LAS i4*)(void*)la, 0, 0);
    __builtin_amdgcn_global_load_async_to_lds_b128(
        (GAS i4*)(void*)gb0, (LAS i4*)(void*)lb0, 0, 0);
    __builtin_amdgcn_global_load_async_to_lds_b128(
        (GAS i4*)(void*)(gb0 + 8), (LAS i4*)(void*)(lb0 + 8), 0, 0);
#else
    *(h8*)la = *(const h8*)ga;
    *(h8*)lb0 = *(const h8*)gb0;
    *(h8*)(lb0 + 8) = *(const h8*)(gb0 + 8);
#endif
  };

  int T = K / 32;
  stage(0, 0);

  for (int t = 0; t < T; ++t) {
    int cur = t & 1;
    if (t + 1 < T) {
      stage(t + 1, cur ^ 1);   // prefetch next tile into other buffer
      ASYNC_WAIT(3);           // tile t complete; next tile may stay in flight
    } else {
      ASYNC_WAIT(0);
    }
    __syncthreads();           // all waves' tile-t data visible

    // A fragment (16x32): lane -> (m, k-group), halfs -> K pattern
    const _Float16* ap = &sA[cur][(wm * 16 + mrow) * LDA_P + half_sel * 8];
    v16h afrag = make_v16h(*(const h8*)ap, *(const h8*)(ap + 16));

#pragma unroll
    for (int q = 0; q < 4; ++q) {
      // B fragment (32x16): lane -> N, halfs -> K (table order)
      v16h bfrag;
#pragma unroll
      for (int s = 0; s < 16; ++s)
        bfrag[s] = sB[cur][((s & 7) + half_sel * 8 + (s >> 3) * 16) * LDB_P + wn * 64 + q * 16 + mrow];
      acc[q] = __builtin_amdgcn_wmma_f32_16x16x32_f16(
          false, afrag, false, bfrag, (short)0, acc[q], false, false);
    }
    __syncthreads();           // readers done before buffer reuse
  }

  int rbase = m0 + wm * 16 + half_sel * 8;
  int cbase = n0 + wn * 64 + (lane & 15);
#pragma unroll
  for (int q = 0; q < 4; ++q) {
    int col = cbase + q * 16;
    float bvv = bias ? bias[col] : 0.0f;
#pragma unroll
    for (int i = 0; i < 8; ++i) {
      int r = rbase + i;
      float v = acc[q][i] + bvv;
      if (resid) v += resid[(long)r * N + col];
      if (do_relu) v = fmaxf(v, 0.0f);
      if (Cf) Cf[(long)r * N + col] = v;
      if (Ch) Ch[(long)r * N + col] = (_Float16)v;
    }
  }
}

// ---------------------------------------------------------------------------
// Flash-style attention: grid (B*H, L/128); block 256 = 8 waves;
// each wave owns 16 query rows; key tiles of 32; online softmax.
// ---------------------------------------------------------------------------
#define SK_P 40   // sK row pad (halfs)
#define SP_P 40   // sP row pad (halfs)

__global__ void __launch_bounds__(256)
attn_kernel(const _Float16* __restrict__ qkv, _Float16* __restrict__ Oh) {
  __shared__ __align__(16) _Float16 sK[64 * SK_P];        // [d][j] transposed K tile
  __shared__ __align__(16) _Float16 sP[8 * 16 * SP_P];    // per-wave P tile [m][j]

  const int RS = 3 * D_;  // qkv row stride (halfs)
  int b = blockIdx.x / H_;
  int h = blockIdx.x % H_;
  int q0 = blockIdx.y * 128;

  int tid  = threadIdx.x;
  int wave = tid >> 5, lane = tid & 31;
  int half_sel = lane >> 4;
  int mrow = lane & 15;

  int qcol = h * DH_;
  int kcol = D_ + h * DH_;
  int vcol = 2 * D_ + h * DH_;

  // Q fragments (two K-chunks of 32 over DH=64), loaded once
  long qrow = (long)(b * L_ + q0 + wave * 16 + mrow) * RS + qcol;
  v16h qf[2];
#pragma unroll
  for (int c = 0; c < 2; ++c) {
    h8 lo = *(const h8*)&qkv[qrow + c * 32 + half_sel * 8];
    h8 hi = *(const h8*)&qkv[qrow + c * 32 + 16 + half_sel * 8];
    qf[c] = make_v16h(lo, hi);
  }

  float Mi[8], Li[8];
#pragma unroll
  for (int i = 0; i < 8; ++i) { Mi[i] = -1e30f; Li[i] = 0.0f; }
  v8f oacc[4] = {};

  _Float16* sPw = &sP[wave * 16 * SP_P];

  int kj = tid >> 3;           // 0..31 : key index in tile
  int dg = tid & 7;            // 0..7  : d-group of 8

  for (int kt = 0; kt < L_ / 32; ++kt) {
    __syncthreads();
    // stage K tile transposed: sK[d][j] = K[kt*32+j][d]
    {
      h8 tmp = *(const h8*)&qkv[(long)(b * L_ + kt * 32 + kj) * RS + kcol + dg * 8];
#pragma unroll
      for (int i = 0; i < 8; ++i) sK[(dg * 8 + i) * SK_P + kj] = tmp[i];
    }
    __syncthreads();

    // S = Q K^T : two 16x16 sub-tiles (j 0..15, 16..31), K-dim = DH = 64
    v8f s0 = {}, s1 = {};
#pragma unroll
    for (int c = 0; c < 2; ++c) {
      v16h b0, b1;
#pragma unroll
      for (int s = 0; s < 16; ++s) {
        int kk = c * 32 + (s & 7) + half_sel * 8 + (s >> 3) * 16;
        b0[s] = sK[kk * SK_P + mrow];
        b1[s] = sK[kk * SK_P + 16 + mrow];
      }
      s0 = __builtin_amdgcn_wmma_f32_16x16x32_f16(false, qf[c], false, b0, (short)0, s0, false, false);
      s1 = __builtin_amdgcn_wmma_f32_16x16x32_f16(false, qf[c], false, b1, (short)0, s1, false, false);
    }

    // online softmax per row (row = i + 8*half_sel, spread over 16-lane group)
#pragma unroll
    for (int i = 0; i < 8; ++i) {
      float a0 = s0[i] * SCALE_;
      float a1 = s1[i] * SCALE_;
      float mx = fmaxf(a0, a1);
      mx = fmaxf(mx, __shfl_xor(mx, 1));
      mx = fmaxf(mx, __shfl_xor(mx, 2));
      mx = fmaxf(mx, __shfl_xor(mx, 4));
      mx = fmaxf(mx, __shfl_xor(mx, 8));
      float mnew = fmaxf(Mi[i], mx);
      float corr = __expf(Mi[i] - mnew);
      float p0 = __expf(a0 - mnew);
      float p1 = __expf(a1 - mnew);
      float ls = p0 + p1;
      ls += __shfl_xor(ls, 1);
      ls += __shfl_xor(ls, 2);
      ls += __shfl_xor(ls, 4);
      ls += __shfl_xor(ls, 8);
      Li[i] = Li[i] * corr + ls;
      Mi[i] = mnew;
#pragma unroll
      for (int t = 0; t < 4; ++t) oacc[t][i] *= corr;
      int m = i + 8 * half_sel;
      sPw[m * SP_P + (lane & 15)]      = (_Float16)p0;
      sPw[m * SP_P + 16 + (lane & 15)] = (_Float16)p1;
    }
    asm volatile("s_wait_dscnt 0" ::: "memory");  // wave-private LDS RAW

    // P fragment (A layout, 16x32)
    const _Float16* pp = &sPw[mrow * SP_P + half_sel * 8];
    v16h pfrag = make_v16h(*(const h8*)pp, *(const h8*)(pp + 16));

    // V fragments straight from global: lane -> d, halfs -> key j (table order)
#pragma unroll
    for (int t = 0; t < 4; ++t) {
      v16h vfrag;
#pragma unroll
      for (int s = 0; s < 16; ++s)
        vfrag[s] = qkv[(long)(b * L_ + kt * 32 + (s & 7) + half_sel * 8 + (s >> 3) * 16) * RS + vcol + t * 16 + mrow];
      oacc[t] = __builtin_amdgcn_wmma_f32_16x16x32_f16(
          false, pfrag, false, vfrag, (short)0, oacc[t], false, false);
    }
  }

  // normalize + store
  int rbase = b * L_ + q0 + wave * 16 + half_sel * 8;
#pragma unroll
  for (int t = 0; t < 4; ++t) {
    int col = h * DH_ + t * 16 + (lane & 15);
#pragma unroll
    for (int i = 0; i < 8; ++i) {
      float v = oacc[t][i] / Li[i];
      Oh[(long)(rbase + i) * D_ + col] = (_Float16)v;
    }
  }
}

// ---------------------------------------------------------------------------
// LayerNorm over D=768 (one block per row)
// ---------------------------------------------------------------------------
__global__ void __launch_bounds__(256)
ln_kernel(const float* __restrict__ in, const float* __restrict__ g,
          const float* __restrict__ bb, float* __restrict__ outf,
          _Float16* __restrict__ outh) {
  long row = blockIdx.x;
  const float* xr = in + row * D_;
  int tid = threadIdx.x, lane = tid & 31, wv = tid >> 5;
  __shared__ float sr[8];

  float s = 0.0f;
  for (int d = tid; d < D_; d += 256) s += xr[d];
  for (int o = 16; o > 0; o >>= 1) s += __shfl_xor(s, o);
  if (lane == 0) sr[wv] = s;
  __syncthreads();
  float mean = 0.0f;
#pragma unroll
  for (int w = 0; w < 8; ++w) mean += sr[w];
  mean *= (1.0f / D_);
  __syncthreads();

  float v = 0.0f;
  for (int d = tid; d < D_; d += 256) { float t = xr[d] - mean; v += t * t; }
  for (int o = 16; o > 0; o >>= 1) v += __shfl_xor(v, o);
  if (lane == 0) sr[wv] = v;
  __syncthreads();
  float var = 0.0f;
#pragma unroll
  for (int w = 0; w < 8; ++w) var += sr[w];
  float inv = rsqrtf(var * (1.0f / D_) + 1e-5f);

  for (int d = tid; d < D_; d += 256) {
    float o = (xr[d] - mean) * inv * g[d] + bb[d];
    if (outf) outf[row * D_ + d] = o;
    if (outh) outh[row * D_ + d] = (_Float16)o;
  }
}

// ---------------------------------------------------------------------------
// q1 = patch_query @ wq.T + bq  (768 outputs, one block each)
// ---------------------------------------------------------------------------
__global__ void __launch_bounds__(256)
q1_kernel(const float* __restrict__ pq, const float* __restrict__ Wq,
          const float* __restrict__ bq, float* __restrict__ q1) {
  int n = blockIdx.x;
  int tid = threadIdx.x, lane = tid & 31, wv = tid >> 5;
  __shared__ float sr[8];
  float s = 0.0f;
  for (int d = tid; d < D_; d += 256) s += pq[d] * Wq[(long)n * D_ + d];
  for (int o = 16; o > 0; o >>= 1) s += __shfl_xor(s, o);
  if (lane == 0) sr[wv] = s;
  __syncthreads();
  if (tid == 0) {
    float t = 0.0f;
    for (int w = 0; w < 8; ++w) t += sr[w];
    q1[n] = t + bq[n];
  }
}

// ---------------------------------------------------------------------------
// Segmented single-query pooling over patch spans (span = 4).
// grid = B*P*H, block = 64 (thread = dh)
// ---------------------------------------------------------------------------
__global__ void __launch_bounds__(64)
pool_kernel(const float* __restrict__ q1, const _Float16* __restrict__ kv,
            const int* __restrict__ ps, const int* __restrict__ pe,
            _Float16* __restrict__ pooled) {
  int gid = blockIdx.x;
  int h = gid % H_;
  int bp = gid / H_;
  int p = bp % P_;
  int b = bp / P_;
  int s0 = ps[b * P_ + p];
  int cnt = pe[b * P_ + p] - s0;
  if (cnt > 16) cnt = 16;
  int tid = threadIdx.x, lane = tid & 31, wv = tid >> 5;
  __shared__ float sred[16][2];

  float qv = q1[h * DH_ + tid];
  for (int li = 0; li < cnt; ++li) {
    float v = qv * (float)kv[(long)(b * L_ + s0 + li) * (2 * D_) + h * DH_ + tid];
    for (int o = 16; o > 0; o >>= 1) v += __shfl_xor(v, o);
    if (lane == 0) sred[li][wv] = v;
  }
  __syncthreads();

  float mx = -1e30f;
  for (int li = 0; li < cnt; ++li) {
    float s = (sred[li][0] + sred[li][1]) * SCALE_;
    mx = fmaxf(mx, s);
  }
  float sm = 0.0f;
  for (int li = 0; li < cnt; ++li)
    sm += __expf((sred[li][0] + sred[li][1]) * SCALE_ - mx);
  float out = 0.0f;
  for (int li = 0; li < cnt; ++li) {
    float w = __expf((sred[li][0] + sred[li][1]) * SCALE_ - mx);
    out += w * (float)kv[(long)(b * L_ + s0 + li) * (2 * D_) + D_ + h * DH_ + tid];
  }
  out /= sm;
  pooled[(long)(b * P_ + p) * D_ + h * DH_ + tid] = (_Float16)out;
}

// ---------------------------------------------------------------------------
extern "C" void kernel_launch(void* const* d_in, const int* in_sizes, int n_in,
                              void* d_out, int out_size, void* d_ws, size_t ws_size,
                              hipStream_t stream) {
  (void)in_sizes; (void)n_in; (void)out_size; (void)ws_size;

  const int*   bytesI    = (const int*)d_in[0];
  const int*   ps        = (const int*)d_in[1];
  const int*   pe        = (const int*)d_in[2];
  const float* byte_emb  = (const float*)d_in[3];
  const float* ngram_tab = (const float*)d_in[4];
  const float* ngram_w   = (const float*)d_in[5];
  const float* ngram_b   = (const float*)d_in[6];
  const float* attn_in_w = (const float*)d_in[7];
  const float* attn_in_b = (const float*)d_in[8];
  const float* attn_out_w= (const float*)d_in[9];
  const float* attn_out_b= (const float*)d_in[10];
  const float* ln1_g     = (const float*)d_in[11];
  const float* ln1_b     = (const float*)d_in[12];
  const float* ffn_w1    = (const float*)d_in[13];
  const float* ffn_b1    = (const float*)d_in[14];
  const float* ffn_w2    = (const float*)d_in[15];
  const float* ffn_b2    = (const float*)d_in[16];
  const float* ln2_g     = (const float*)d_in[17];
  const float* ln2_b     = (const float*)d_in[18];
  const float* pqv       = (const float*)d_in[19];
  const float* pa_in_w   = (const float*)d_in[20];
  const float* pa_in_b   = (const float*)d_in[21];
  const float* pa_out_w  = (const float*)d_in[22];
  const float* pa_out_b  = (const float*)d_in[23];

  char* ws = (char*)d_ws;
  size_t off = 0;
  auto alloc = [&](size_t nbytes) -> char* {
    char* pp = ws + off;
    off = (off + nbytes + 255) & ~(size_t)255;
    return pp;
  };

  _Float16* wt_ngram = (_Float16*)alloc((size_t)4608 * 768 * 2);
  _Float16* wt_qkv   = (_Float16*)alloc((size_t)768 * 2304 * 2);
  _Float16* wt_ao    = (_Float16*)alloc((size_t)768 * 768 * 2);
  _Float16* wt_f1    = (_Float16*)alloc((size_t)768 * 3072 * 2);
  _Float16* wt_f2    = (_Float16*)alloc((size_t)3072 * 768 * 2);
  _Float16* wt_kv    = (_Float16*)alloc((size_t)768 * 1536 * 2);
  _Float16* wt_po    = (_Float16*)alloc((size_t)768 * 768 * 2);

  float*    x0      = (float*)   alloc((size_t)BL_ * D_ * 4);
  _Float16* cat     = (_Float16*)alloc((size_t)BL_ * 6 * D_ * 2);
  float*    xf      = (float*)   alloc((size_t)BL_ * D_ * 4);
  _Float16* xh      = (_Float16*)alloc((size_t)BL_ * D_ * 2);
  _Float16* qkvh    = (_Float16*)alloc((size_t)BL_ * 3 * D_ * 2);
  _Float16* oh      = (_Float16*)alloc((size_t)BL_ * D_ * 2);
  float*    y1      = (float*)   alloc((size_t)BL_ * D_ * 4);
  float*    x1      = (float*)   alloc((size_t)BL_ * D_ * 4);
  _Float16* x1h     = (_Float16*)alloc((size_t)BL_ * D_ * 2);
  _Float16* ffnh    = (_Float16*)alloc((size_t)BL_ * 4 * D_ * 2);
  float*    y2      = (float*)   alloc((size_t)BL_ * D_ * 4);
  _Float16* ench    = (_Float16*)alloc((size_t)BL_ * D_ * 2);
  _Float16* kvh     = (_Float16*)alloc((size_t)BL_ * 2 * D_ * 2);
  float*    q1w     = (float*)   alloc((size_t)D_ * 4);
  _Float16* pooledh = (_Float16*)alloc((size_t)(B_ * P_) * D_ * 2);

  auto cvt = [&](const float* W, _Float16* Wt, int N, int K) {
    long tot = (long)N * K;
    cvt_t_kernel<<<dim3((unsigned)((tot + 255) / 256)), dim3(256), 0, stream>>>(W, Wt, N, K);
  };
  cvt(ngram_w,              wt_ngram, 768,  4608);
  cvt(attn_in_w,            wt_qkv,   2304, 768);
  cvt(attn_out_w,           wt_ao,    768,  768);
  cvt(ffn_w1,               wt_f1,    3072, 768);
  cvt(ffn_w2,               wt_f2,    768,  3072);
  cvt(pa_in_w + 768 * 768,  wt_kv,    1536, 768);   // [wk; wv] rows
  cvt(pa_out_w,             wt_po,    768,  768);

  gather_kernel<<<dim3(BL_), dim3(256), 0, stream>>>(bytesI, byte_emb, ngram_tab, x0, cat);

  auto gemm = [&](const _Float16* A, const _Float16* Bt, const float* bias,
                  const float* resid, float* Cf, _Float16* Ch,
                  int M, int N, int K, int relu) {
    gemm_wmma_kernel<<<dim3((unsigned)(N / 128), (unsigned)(M / 64)), dim3(256), 0, stream>>>(
        A, Bt, bias, resid, Cf, Ch, M, N, K, relu);
  };

  // x = x0 + cat @ Wn^T + b
  gemm(cat, wt_ngram, ngram_b, x0, xf, xh, BL_, 768, 4608, 0);
  // qkv
  gemm(xh, wt_qkv, attn_in_b, nullptr, nullptr, qkvh, BL_, 2304, 768, 0);
  // attention
  attn_kernel<<<dim3(B_ * H_, L_ / 128), dim3(256), 0, stream>>>(qkvh, oh);
  // attn out proj + residual
  gemm(oh, wt_ao, attn_out_b, xf, y1, nullptr, BL_, 768, 768, 0);
  // LN1
  ln_kernel<<<dim3(BL_), dim3(256), 0, stream>>>(y1, ln1_g, ln1_b, x1, x1h);
  // FFN
  gemm(x1h, wt_f1, ffn_b1, nullptr, nullptr, ffnh, BL_, 3072, 768, 1);
  gemm(ffnh, wt_f2, ffn_b2, x1, y2, nullptr, BL_, 768, 3072, 0);
  // LN2 -> enc (f16 only)
  ln_kernel<<<dim3(BL_), dim3(256), 0, stream>>>(y2, ln2_g, ln2_b, nullptr, ench);
  // kk|vv projection
  gemm(ench, wt_kv, pa_in_b + 768, nullptr, nullptr, kvh, BL_, 1536, 768, 0);
  // q1
  q1_kernel<<<dim3(768), dim3(256), 0, stream>>>(pqv, pa_in_w, pa_in_b, q1w);
  // patch pooling
  pool_kernel<<<dim3(B_ * P_ * H_), dim3(64), 0, stream>>>(q1w, kvh, ps, pe, pooledh);
  // final projection -> d_out (f32)
  gemm(pooledh, wt_po, pa_out_b, nullptr, (float*)d_out, nullptr, B_ * P_, 768, 768, 0);
}
